// FusionModel_81183471829694
// MI455X (gfx1250) — compile-verified, matrix operand order checked by reading, not run on hardware
//
#include <hip/hip_runtime.h>
#include <math.h>

typedef __attribute__((ext_vector_type(16))) __bf16 v16bf;
typedef __attribute__((ext_vector_type(8)))  float  v8f;
typedef __attribute__((ext_vector_type(2)))  __bf16 bf16x2;
typedef __attribute__((ext_vector_type(2)))  float  f32x2;

union FragBF { v16bf v; uint4 q[2]; };

// fp32 pair -> packed bf16 (RNE) via native v_cvt_pk_bf16_f32 path
__device__ __forceinline__ unsigned int pk_bf16(float a, float b) {
    f32x2 f = {a, b};
    bf16x2 h = __builtin_convertvector(f, bf16x2);
    return __builtin_bit_cast(unsigned int, h);
}

__device__ __forceinline__ float gelu_exact(float x) {
    return 0.5f * x * (1.0f + erff(x * 0.70710678118654752440f));
}

// ---------------------------------------------------------------------------
// Generic bf16-WMMA GEMM:  C[z][m][n] = act(alpha * sum_k A[z][m][k]*Wt[z][n][k]
//                                           + bias + residual)
// A: [M,K] row-major (row stride Ars, batch stride Abs), fp32
// Wt: [N,K] row-major (i.e. B transposed), fp32
// biasMode: 0 none, 1 per-col (bias[n]), 2 per-row (bias[m]); biasBs = batch stride
// storeT: 0 -> C[m*Crs+n], 1 -> C[n*Crs+m]
// Block: 128 threads (4 waves), 64x64 tile, K stepped by 32.
// ---------------------------------------------------------------------------
__global__ __launch_bounds__(128)
void gemm_wmma_bf16(const float* __restrict__ A, long long Abs, int Ars,
                    const float* __restrict__ Wt, long long Wbs, int Wrs,
                    const float* __restrict__ bias, long long biasBs, int biasMode,
                    const float* __restrict__ Res, long long Rbs, int Rrs,
                    float* __restrict__ C, long long Cbs, int Crs,
                    int M, int N, int K, float alpha, int act, int storeT)
{
    __shared__ unsigned short sA[64][48];   // 64 rows x 32 bf16, padded to 48 (96B rows)
    __shared__ unsigned short sW[64][48];

    const int z = blockIdx.z;
    const float* Ab = A + (long long)z * Abs;
    const float* Wb = Wt + (long long)z * Wbs;
    const int m0 = blockIdx.y * 64, n0 = blockIdx.x * 64;
    const int tid = threadIdx.x;
    const int lane = tid & 31, wv = tid >> 5;
    const int lh = lane & 15;               // M (or N) within 16-tile
    const int ks = (lane >> 4) << 3;        // K sub-offset 0 / 8
    const int wm = (wv >> 1) * 32, wn = (wv & 1) * 32;

    v8f acc[2][2];
    #pragma unroll
    for (int i = 0; i < 2; i++)
        #pragma unroll
        for (int j = 0; j < 2; j++) {
            v8f zv = {0.f,0.f,0.f,0.f,0.f,0.f,0.f,0.f};
            acc[i][j] = zv;
        }

    const int r = tid >> 1, half = tid & 1; // 64 rows x 2 k-halves of 16

    for (int k0 = 0; k0 < K; k0 += 32) {
        __syncthreads();
        {
            const int gk = k0 + half * 16;
            // ---- A tile ----
            {
                uint4 w0, w1;
                const int gr = m0 + r;
                if (gr < M) {
                    const float4* s = (const float4*)(Ab + (long long)gr * Ars + gk);
                    float4 f0 = s[0], f1 = s[1], f2 = s[2], f3 = s[3];
                    w0.x = pk_bf16(f0.x, f0.y); w0.y = pk_bf16(f0.z, f0.w);
                    w0.z = pk_bf16(f1.x, f1.y); w0.w = pk_bf16(f1.z, f1.w);
                    w1.x = pk_bf16(f2.x, f2.y); w1.y = pk_bf16(f2.z, f2.w);
                    w1.z = pk_bf16(f3.x, f3.y); w1.w = pk_bf16(f3.z, f3.w);
                } else {
                    w0.x = w0.y = w0.z = w0.w = 0u;
                    w1 = w0;
                }
                *(uint4*)&sA[r][half * 16]     = w0;
                *(uint4*)&sA[r][half * 16 + 8] = w1;
            }
            // ---- W tile ----
            {
                uint4 w0, w1;
                const int gn = n0 + r;
                if (gn < N) {
                    const float4* s = (const float4*)(Wb + (long long)gn * Wrs + gk);
                    float4 f0 = s[0], f1 = s[1], f2 = s[2], f3 = s[3];
                    w0.x = pk_bf16(f0.x, f0.y); w0.y = pk_bf16(f0.z, f0.w);
                    w0.z = pk_bf16(f1.x, f1.y); w0.w = pk_bf16(f1.z, f1.w);
                    w1.x = pk_bf16(f2.x, f2.y); w1.y = pk_bf16(f2.z, f2.w);
                    w1.z = pk_bf16(f3.x, f3.y); w1.w = pk_bf16(f3.z, f3.w);
                } else {
                    w0.x = w0.y = w0.z = w0.w = 0u;
                    w1 = w0;
                }
                *(uint4*)&sW[r][half * 16]     = w0;
                *(uint4*)&sW[r][half * 16 + 8] = w1;
            }
        }
        __syncthreads();

        #pragma unroll
        for (int i = 0; i < 2; i++) {
            FragBF fa;
            const int ar = wm + i * 16 + lh;
            fa.q[0] = *(const uint4*)&sA[ar][ks];        // K = ks..ks+7
            fa.q[1] = *(const uint4*)&sA[ar][16 + ks];   // K = 16+ks..16+ks+7
            #pragma unroll
            for (int j = 0; j < 2; j++) {
                FragBF fb;
                const int br = wn + j * 16 + lh;
                fb.q[0] = *(const uint4*)&sW[br][ks];
                fb.q[1] = *(const uint4*)&sW[br][16 + ks];
                acc[i][j] = __builtin_amdgcn_wmma_f32_16x16x32_bf16(
                    false, fa.v, false, fb.v, (short)0, acc[i][j], false, false);
            }
        }
    }

    float* Cb = C + (long long)z * Cbs;
    const float* Rb = Res ? (Res + (long long)z * Rbs) : (const float*)0;
    const float* bb = bias ? (bias + (long long)z * biasBs) : (const float*)0;
    #pragma unroll
    for (int i = 0; i < 2; i++) {
        #pragma unroll
        for (int j = 0; j < 2; j++) {
            #pragma unroll
            for (int v = 0; v < 8; v++) {
                const int gm = m0 + wm + i * 16 + ks + v;  // C layout: M = v + 8*(lane>=16)
                const int gn = n0 + wn + j * 16 + lh;      // N = lane & 15
                if (gm < M && gn < N) {
                    float val = alpha * acc[i][j][v];
                    if (bb) val += (biasMode == 2) ? bb[gm] : bb[gn];
                    if (Rb) val += Rb[(long long)gm * Rrs + gn];
                    if (act == 1) val = gelu_exact(val);
                    if (storeT) Cb[(long long)gn * Crs + gm] = val;
                    else        Cb[(long long)gm * Crs + gn] = val;
                }
            }
        }
    }
}

// ---------------------------------------------------------------------------
// per-(b,c) stats over L: mean, std(ddof=1), max, min
// ---------------------------------------------------------------------------
__global__ __launch_bounds__(256)
void stats_kernel(const float* __restrict__ x, float* __restrict__ stats, int L, int C)
{
    const int c = blockIdx.x, b = blockIdx.y, t = threadIdx.x;
    const float* xp = x + (long long)b * L * C + c;
    float s = 0.f, s2 = 0.f, mx = -3.4028235e38f, mn = 3.4028235e38f;
    for (int l = t; l < L; l += 256) {
        float v = xp[(long long)l * C];
        s += v; s2 += v * v; mx = fmaxf(mx, v); mn = fminf(mn, v);
    }
    __shared__ float sh0[256], sh1[256], sh2[256], sh3[256];
    sh0[t] = s; sh1[t] = s2; sh2[t] = mx; sh3[t] = mn;
    __syncthreads();
    for (int o = 128; o > 0; o >>= 1) {
        if (t < o) {
            sh0[t] += sh0[t + o]; sh1[t] += sh1[t + o];
            sh2[t] = fmaxf(sh2[t], sh2[t + o]); sh3[t] = fminf(sh3[t], sh3[t + o]);
        }
        __syncthreads();
    }
    if (t == 0) {
        float mean = sh0[0] / (float)L;
        float var  = (sh1[0] - (float)L * mean * mean) / (float)(L - 1);
        float* sp = stats + (long long)b * 4 * C;
        sp[c]         = mean;
        sp[C + c]     = sqrtf(fmaxf(var, 0.f));
        sp[2 * C + c] = sh2[0];
        sp[3 * C + c] = sh3[0];
    }
}

// LayerNorm: grid(rows_per_batch, nbatch); row addr = y*bs + x*W. In-place safe.
__global__ __launch_bounds__(256)
void ln_kernel(const float* __restrict__ in, float* __restrict__ outp,
               const float* __restrict__ g, const float* __restrict__ bvec,
               int W, long long in_bs, long long out_bs)
{
    const float* ip = in   + (long long)blockIdx.y * in_bs  + (long long)blockIdx.x * W;
    float*       op = outp + (long long)blockIdx.y * out_bs + (long long)blockIdx.x * W;
    const int t = threadIdx.x;
    float s = 0.f, s2 = 0.f;
    for (int j = t; j < W; j += 256) { float v = ip[j]; s += v; s2 += v * v; }
    __shared__ float sh0[256], sh1[256];
    sh0[t] = s; sh1[t] = s2; __syncthreads();
    for (int o = 128; o > 0; o >>= 1) {
        if (t < o) { sh0[t] += sh0[t + o]; sh1[t] += sh1[t + o]; }
        __syncthreads();
    }
    const float mean = sh0[0] / (float)W;
    const float var  = sh1[0] / (float)W - mean * mean;
    const float inv  = rsqrtf(var + 1e-5f);
    for (int j = t; j < W; j += 256)
        op[j] = (ip[j] - mean) * inv * g[j] + bvec[j];
}

// softmax over last dim; in-place safe
__global__ __launch_bounds__(256)
void softmax_rows_kernel(const float* __restrict__ in, float* __restrict__ outp, int W)
{
    const long long row = blockIdx.x;
    const float* ip = in + row * (long long)W;
    float*       op = outp + row * (long long)W;
    const int t = threadIdx.x;
    __shared__ float sh[256];
    float mx = -3.4028235e38f;
    for (int j = t; j < W; j += 256) mx = fmaxf(mx, ip[j]);
    sh[t] = mx; __syncthreads();
    for (int o = 128; o > 0; o >>= 1) { if (t < o) sh[t] = fmaxf(sh[t], sh[t + o]); __syncthreads(); }
    mx = sh[0]; __syncthreads();
    float sum = 0.f;
    for (int j = t; j < W; j += 256) { float e = expf(ip[j] - mx); op[j] = e; sum += e; }
    sh[t] = sum; __syncthreads();
    for (int o = 128; o > 0; o >>= 1) { if (t < o) sh[t] += sh[t + o]; __syncthreads(); }
    const float inv = 1.f / sh[0];
    for (int j = t; j < W; j += 256) op[j] *= inv;
}

// softmax over axis 1 (N tokens) of logits (B,N,S)
__global__ __launch_bounds__(256)
void softmax_dim1_kernel(const float* __restrict__ in, float* __restrict__ outp, int N, int S)
{
    const int s = blockIdx.x, b = blockIdx.y, t = threadIdx.x;
    const float* ip = in   + (long long)b * N * S + s;
    float*       op = outp + (long long)b * N * S + s;
    __shared__ float sh[256];
    float mx = -3.4028235e38f;
    for (int n = t; n < N; n += 256) mx = fmaxf(mx, ip[(long long)n * S]);
    sh[t] = mx; __syncthreads();
    for (int o = 128; o > 0; o >>= 1) { if (t < o) sh[t] = fmaxf(sh[t], sh[t + o]); __syncthreads(); }
    mx = sh[0]; __syncthreads();
    float sum = 0.f;
    for (int n = t; n < N; n += 256) { float e = expf(ip[(long long)n * S] - mx); op[(long long)n * S] = e; sum += e; }
    sh[t] = sum; __syncthreads();
    for (int o = 128; o > 0; o >>= 1) { if (t < o) sh[t] += sh[t + o]; __syncthreads(); }
    const float inv = 1.f / sh[0];
    for (int n = t; n < N; n += 256) op[(long long)n * S] *= inv;
}

__global__ void transpose_kernel(const float* __restrict__ in, float* __restrict__ outp,
                                 int R, int Ccols)
{
    int idx = blockIdx.x * blockDim.x + threadIdx.x;
    if (idx < R * Ccols) {
        int rr = idx / Ccols, cc = idx - rr * Ccols;
        outp[(long long)cc * R + rr] = in[idx];
    }
}

// slots[e][b*SL+sl][d] = sum_n disp[b,n,e*SL+sl] * packed[b,n,d]
__global__ __launch_bounds__(256)
void slots_in_kernel(const float* __restrict__ disp, const float* __restrict__ packed,
                     float* __restrict__ slots, int N, int S, int D, int SL, int B)
{
    const int s = blockIdx.x, b = blockIdx.y, d = threadIdx.x;
    const float* dp = disp + (long long)b * N * S + s;
    const float* pp = packed + (long long)b * N * D + d;
    float acc = 0.f;
    for (int n = 0; n < N; n++) acc += dp[(long long)n * S] * pp[(long long)n * D];
    const int e = s / SL, sl = s - e * SL;
    slots[(((long long)e * B + b) * SL + sl) * D + d] = acc;
}

// out[b,n,d] = packed[b,n,d] + sum_s comb[b,n,s] * slots[e][b*SL+sl][d]
__global__ __launch_bounds__(256)
void moe_out_kernel(const float* __restrict__ comb, const float* __restrict__ slots,
                    const float* __restrict__ packed, float* __restrict__ outp,
                    int N, int S, int D, int SL, int B)
{
    const int n = blockIdx.x, b = blockIdx.y, d = threadIdx.x;
    float acc = packed[((long long)b * N + n) * D + d];
    const float* cp = comb + ((long long)b * N + n) * S;
    for (int s = 0; s < S; s++) {
        const int e = s / SL, sl = s - e * SL;
        acc += cp[s] * slots[(((long long)e * B + b) * SL + sl) * D + d];
    }
    outp[((long long)b * N + n) * D + d] = acc;
}

// ---------------------------------------------------------------------------
extern "C" void kernel_launch(void* const* d_in, const int* in_sizes, int n_in,
                              void* d_out, int out_size, void* d_ws, size_t ws_size,
                              hipStream_t stream)
{
    (void)in_sizes; (void)n_in; (void)out_size;

    const float* x        = (const float*)d_in[0];
    const float* h1       = (const float*)d_in[1];
    const float* h2       = (const float*)d_in[2];
    const float* proj1_w  = (const float*)d_in[3];
    const float* proj1_b  = (const float*)d_in[4];
    const float* proj1_g  = (const float*)d_in[5];
    const float* proj1_bn = (const float*)d_in[6];
    const float* proj2_w  = (const float*)d_in[7];
    const float* proj2_b  = (const float*)d_in[8];
    const float* proj2_g  = (const float*)d_in[9];
    const float* proj2_bn = (const float*)d_in[10];
    const float* qg_w1    = (const float*)d_in[11];
    const float* qg_b1    = (const float*)d_in[12];
    const float* qg_w2    = (const float*)d_in[13];
    const float* qg_b2    = (const float*)d_in[14];
    const float* qg_g     = (const float*)d_in[15];
    const float* qg_bn    = (const float*)d_in[16];
    const float* attn_in_w  = (const float*)d_in[17];
    const float* attn_in_b  = (const float*)d_in[18];
    const float* attn_out_w = (const float*)d_in[19];
    const float* attn_out_b = (const float*)d_in[20];
    const float* ln1_g    = (const float*)d_in[21];
    const float* ln1_b    = (const float*)d_in[22];
    const float* phi      = (const float*)d_in[23];
    const float* exp_w1   = (const float*)d_in[24];
    const float* exp_b1   = (const float*)d_in[25];
    const float* exp_w2   = (const float*)d_in[26];
    const float* exp_b2   = (const float*)d_in[27];
    const float* ln2_g    = (const float*)d_in[28];
    const float* ln2_b    = (const float*)d_in[29];
    const float* head_w   = (const float*)d_in[30];
    const float* head_b   = (const float*)d_in[31];
    const float* agg_w    = (const float*)d_in[32];
    const float* agg_b    = (const float*)d_in[33];
    float* outp = (float*)d_out;
    float* ws   = (float*)d_ws;

    const int B = 128, L = 1024, C = 64, Q = 256, D = 256, DH = 1024;
    const int E = 8, SL = 4, P = 96, S = 32, H = 8, KV = 128;
    const long long QD = (long long)Q * D;          // 65536
    const long long SZ_KV = (long long)B * KV * D;  // 4.19M
    const long long SZ_QD = (long long)B * QD;      // 8.39M

    // -------- scratch arena (lifetime-based aliasing), ~231 MB --------
    float* kv      = ws;                                   // (B,128,D) t1|t2, LN'd in place
    float* qbuf    = kv + SZ_KV;                           // q (B,Q,D)
    float* qp      = qbuf + SZ_QD;                         // q-proj; aliases qraw & pkraw
    float* qraw    = qp;                                   // pre-LN qflat (dead before qp)
    float* pkraw   = qp;                                   // pre-LN packed (qp dead by then)
    float* kp      = qp + SZ_QD;                           // (B,128,D)
    float* vpT     = kp + SZ_KV;                           // (B,D,128) V transposed
    float* ctx     = vpT + SZ_KV;                          // (B,Q,D); aliases moeraw
    float* moeraw  = ctx;
    float* fused   = kp;                                   // aliases kp+vpT (dead after attn)
    float* scoresb = ctx + SZ_QD;                          // (B,Q,128) one head; aliases outT
    float* outT    = scoresb;                              // (B,P,Q)
    float* packed  = scoresb + (long long)B * Q * KV;      // (B,Q,D)
    float* logits  = packed + SZ_QD;                       // (B,Q,S)
    float* disp    = logits + (long long)B * Q * S;
    float* comb    = disp + (long long)B * Q * S;
    float* slotsin = comb + (long long)B * Q * S;          // (E,B*SL,D)
    float* hmid    = slotsin + (long long)E * B * SL * D;  // (E,B*SL,2D)
    float* slotso  = hmid + (long long)E * B * SL * 2 * D; // (E,B*SL,D)
    float* statsb  = slotso + (long long)E * B * SL * D;   // (B,4C)
    float* gbuf    = statsb + (long long)B * 4 * C;        // (B,D)
    float* phiT    = gbuf + (long long)B * D;              // (S,D)
    const long long need = (phiT - ws) + (long long)S * D;
    if (ws_size < (size_t)need * sizeof(float)) return;    // scratch too small: bail

    auto gemm = [&](const float* A, long long Abs, int Ars,
                    const float* W, long long Wbs, int Wrs,
                    const float* bias, long long bBs, int bMode,
                    const float* R, long long Rbs, int Rrs,
                    float* Cp, long long Cbs, int Crs,
                    int M, int N, int K, float alpha, int act, int storeT, int Z) {
        dim3 grid((N + 63) / 64, (M + 63) / 64, Z);
        gemm_wmma_bf16<<<grid, dim3(128), 0, stream>>>(
            A, Abs, Ars, W, Wbs, Wrs, bias, bBs, bMode,
            R, Rbs, Rrs, Cp, Cbs, Crs, M, N, K, alpha, act, storeT);
    };

    // phi (D,S) -> phiT (S,D)
    transpose_kernel<<<(D * S + 255) / 256, 256, 0, stream>>>(phi, phiT, D, S);
    // stats over L for each (b,c)
    stats_kernel<<<dim3(C, B), 256, 0, stream>>>(x, statsb, L, C);

    // t1 = h1 @ proj1_w.T + b -> kv[:, :64, :] ; t2 -> kv[:, 64:, :]
    gemm(h1, (long long)C * DH, DH, proj1_w, 0, DH, proj1_b, 0, 1, nullptr, 0, 0,
         kv,              (long long)KV * D, D, C, D, DH, 1.f, 0, 0, B);
    gemm(h2, (long long)C * DH, DH, proj2_w, 0, DH, proj2_b, 0, 1, nullptr, 0, 0,
         kv + 64LL * D,   (long long)KV * D, D, C, D, DH, 1.f, 0, 0, B);
    ln_kernel<<<dim3(64, B), 256, 0, stream>>>(kv,            kv,            proj1_g, proj1_bn, D, (long long)KV * D, (long long)KV * D);
    ln_kernel<<<dim3(64, B), 256, 0, stream>>>(kv + 64LL * D, kv + 64LL * D, proj2_g, proj2_bn, D, (long long)KV * D, (long long)KV * D);

    // g = gelu(stats @ qg_w1.T + b1)        (M=B, single GEMM)
    gemm(statsb, 0, 4 * C, qg_w1, 0, 4 * C, qg_b1, 0, 1, nullptr, 0, 0,
         gbuf, 0, D, B, D, 4 * C, 1.f, 1, 0, 1);
    // qraw = g @ qg_w2.T + b2 ; LN over full 65536 axis -> q
    gemm(gbuf, 0, D, qg_w2, 0, D, qg_b2, 0, 1, nullptr, 0, 0,
         qraw, 0, (int)QD, B, (int)QD, D, 1.f, 0, 0, 1);
    ln_kernel<<<dim3(1, B), 256, 0, stream>>>(qraw, qbuf, qg_g, qg_bn, (int)QD, QD, QD);

    // Q/K/V projections (V stored transposed per batch: vpT[b][d][k])
    gemm(qbuf, QD, D, attn_in_w,               0, D, attn_in_b,           0, 1, nullptr, 0, 0,
         qp,  QD,               D, Q,  D, D, 1.f, 0, 0, B);
    gemm(kv, (long long)KV * D, D, attn_in_w + (long long)D * D,     0, D, attn_in_b + D,     0, 1, nullptr, 0, 0,
         kp,  (long long)KV * D, D, KV, D, D, 1.f, 0, 0, B);
    gemm(kv, (long long)KV * D, D, attn_in_w + 2LL * D * D, 0, D, attn_in_b + 2 * D, 0, 1, nullptr, 0, 0,
         vpT, (long long)D * KV, KV, KV, D, D, 1.f, 0, 1, B);

    // attention, per head (keeps score scratch at 16 MB)
    const float scl = 0.17677669529663688f;   // 1/sqrt(32)
    for (int h = 0; h < H; h++) {
        gemm(qp + h * 32, QD, D, kp + h * 32, (long long)KV * D, D, nullptr, 0, 0, nullptr, 0, 0,
             scoresb, (long long)Q * KV, KV, Q, KV, 32, scl, 0, 0, B);
        softmax_rows_kernel<<<dim3(B * Q), 256, 0, stream>>>(scoresb, scoresb, KV);
        gemm(scoresb, (long long)Q * KV, KV, vpT + (long long)h * 32 * KV, (long long)D * KV, KV,
             nullptr, 0, 0, nullptr, 0, 0,
             ctx + h * 32, QD, D, Q, 32, KV, 1.f, 0, 0, B);
    }

    // attn_out + residual(q), LN -> packed
    gemm(ctx, QD, D, attn_out_w, 0, D, attn_out_b, 0, 1, qbuf, QD, D,
         pkraw, QD, D, Q, D, D, 1.f, 0, 0, B);
    ln_kernel<<<dim3(Q, B), 256, 0, stream>>>(pkraw, packed, ln1_g, ln1_b, D, QD, QD);

    // logits = packed @ phi  (via phiT as [S,D])
    gemm(packed, QD, D, phiT, 0, D, nullptr, 0, 0, nullptr, 0, 0,
         logits, (long long)Q * S, S, Q, S, D, 1.f, 0, 0, B);
    softmax_dim1_kernel<<<dim3(S, B), 256, 0, stream>>>(logits, disp, Q, S);  // dispatch
    softmax_rows_kernel<<<dim3(B * Q), 256, 0, stream>>>(logits, comb, S);    // combine

    // MoE
    slots_in_kernel<<<dim3(S, B), 256, 0, stream>>>(disp, packed, slotsin, Q, S, D, SL, B);
    gemm(slotsin, (long long)B * SL * D, D, exp_w1, 2LL * D * D, D, exp_b1, 2 * D, 1, nullptr, 0, 0,
         hmid, (long long)B * SL * 2 * D, 2 * D, B * SL, 2 * D, D, 1.f, 1, 0, E);
    gemm(hmid, (long long)B * SL * 2 * D, 2 * D, exp_w2, 2LL * D * D, 2 * D, exp_b2, D, 1, nullptr, 0, 0,
         slotso, (long long)B * SL * D, D, B * SL, D, 2 * D, 1.f, 0, 0, E);
    moe_out_kernel<<<dim3(Q, B), 256, 0, stream>>>(comb, slotso, packed, moeraw, Q, S, D, SL, B);
    ln_kernel<<<dim3(Q, B), 256, 0, stream>>>(moeraw, fused, ln2_g, ln2_b, D, QD, QD);

    // head (store transposed: outT[b][p][q])
    gemm(fused, QD, D, head_w, 0, D, head_b, 0, 1, nullptr, 0, 0,
         outT, (long long)P * Q, Q, Q, P, D, 1.f, 0, 1, B);
    // agg: A=agg_w [C,Q], Wt=outT[b] [P,Q], per-row bias agg_b[f], transposed store
    // -> final[b][p][f]  (== agg.transpose(0,2,1))
    gemm(agg_w, 0, Q, outT, (long long)P * Q, Q, agg_b, 0, 2, nullptr, 0, 0,
         outp, (long long)P * C, C, C, P, Q, 1.f, 0, 1, B);
}
